// MoERouterCoeff_3805341024606
// MI455X (gfx1250) — compile-verified, matrix operand order checked by reading, not run on hardware
//
#include <hip/hip_runtime.h>
#include <hip/hip_bf16.h>

typedef __attribute__((ext_vector_type(16))) __bf16 v16bf;
typedef __attribute__((ext_vector_type(8)))  __bf16 v8bf;
typedef __attribute__((ext_vector_type(8)))  float  v8f;

#define NROWS  16384
#define IN_DIM 1024
#define HID    4096
#define NGRP   6
#define POOL   64
#define OUTD   (NGRP * POOL)   // 384
#define TOPK   2

__device__ __forceinline__ unsigned pk_bf16(float a, float b) {
  unsigned short x = __builtin_bit_cast(unsigned short, (__bf16)a);
  unsigned short y = __builtin_bit_cast(unsigned short, (__bf16)b);
  return ((unsigned)y << 16) | (unsigned)x;
}

__device__ __forceinline__ float fast_tanh(float u) {
  // tanh(u) = 1 - 2/(1+exp(2u)); saturates correctly at +/-inf
  return 1.0f - 2.0f / (1.0f + __expf(2.0f * u));
}
__device__ __forceinline__ float gelu_tanh(float x) {
  float u = 0.7978845608028654f * (x + 0.044715f * x * x * x);
  return 0.5f * x * (1.0f + fast_tanh(u));
}
__device__ __forceinline__ float sig2(float x) {
  return 2.0f / (1.0f + __expf(-x));
}

// C = act(A[M,K] @ Bw[K,N] + bias). BM=BN=128, BK=32, 256 threads (8 waves),
// double-buffered LDS (ping/pong), one barrier per K-step.
// AT: float (converted to bf16 on LDS stage) or __bf16 (copied directly).
// ACT: 0 -> GELU(tanh) with OT=__bf16 ; 1 -> 2*sigmoid with OT=float.
template <typename AT, int ACT, typename OT>
__global__ __launch_bounds__(256) void gemm_bias_act(
    const AT* __restrict__ A, const float* __restrict__ Bw,
    const float* __restrict__ bias, OT* __restrict__ out,
    int M, int N, int K)
{
  __shared__ __bf16 la[2][128 * 40];  // [buf][row][k], row stride 40 (80B, 16B aligned)
  __shared__ __bf16 lb[2][128 * 40];  // [buf][col][k] (B transposed)

  const int tid   = threadIdx.x;
  const int lane  = tid & 31;
  const int wid   = tid >> 5;
  const int waveM = (wid & 3) << 5;   // 0,32,64,96
  const int waveN = (wid >> 2) << 6;  // 0,64
  const int m0 = blockIdx.y * 128;
  const int n0 = blockIdx.x * 128;

  v8f acc[2][4];
  #pragma unroll
  for (int tm = 0; tm < 2; ++tm)
    #pragma unroll
    for (int tn = 0; tn < 4; ++tn) {
      v8f z = {0.f, 0.f, 0.f, 0.f, 0.f, 0.f, 0.f, 0.f};
      acc[tm][tn] = z;
    }

  const int ml  = lane & 15;          // row (A) / col (B) within 16x16 tile
  const int khf = (lane >> 4) << 3;   // k-half select: 0 or 8 (bf16 elements)

  // B-stage mapping: thread -> (column bn, k-half bkh)
  const int bn  = tid & 127;
  const int bkh = tid >> 7;           // 0 -> k 0..15, 1 -> k 16..31

  const int T = K >> 5;               // number of 32-deep K tiles

  // register staging for global -> LDS double buffering
  float4 aF[4];                       // used when AT == float
  uint4  aH[2];                       // used when AT == __bf16
  float  bF[16];

  auto load_tiles = [&](int kt) {
    const int k0 = kt << 5;
    if constexpr (sizeof(AT) == 4) {
      #pragma unroll
      for (int i = 0; i < 4; ++i) {
        int idx = tid + (i << 8);
        int r = idx >> 3, c = idx & 7;
        aF[i] = *(const float4*)(A + (size_t)(m0 + r) * K + k0 + (c << 2));
      }
    } else {
      #pragma unroll
      for (int i = 0; i < 2; ++i) {
        int idx = tid + (i << 8);
        int r = idx >> 2, c = idx & 3;
        aH[i] = *(const uint4*)(A + (size_t)(m0 + r) * K + k0 + (c << 3));
      }
    }
    const float* bp = Bw + (size_t)(k0 + (bkh << 4)) * N + n0 + bn;
    #pragma unroll
    for (int j = 0; j < 16; ++j) bF[j] = bp[(size_t)j * N];  // coalesced across lanes
  };

  auto store_tiles = [&](int buf) {
    __bf16* La = la[buf];
    __bf16* Lb = lb[buf];
    if constexpr (sizeof(AT) == 4) {
      #pragma unroll
      for (int i = 0; i < 4; ++i) {
        int idx = tid + (i << 8);
        int r = idx >> 3, c = idx & 7;
        uint2 u = make_uint2(pk_bf16(aF[i].x, aF[i].y), pk_bf16(aF[i].z, aF[i].w));
        *(uint2*)&La[r * 40 + (c << 2)] = u;
      }
    } else {
      #pragma unroll
      for (int i = 0; i < 2; ++i) {
        int idx = tid + (i << 8);
        int r = idx >> 2, c = idx & 3;
        *(uint4*)&La[r * 40 + (c << 3)] = aH[i];
      }
    }
    uint4 u0 = make_uint4(pk_bf16(bF[0], bF[1]),   pk_bf16(bF[2], bF[3]),
                          pk_bf16(bF[4], bF[5]),   pk_bf16(bF[6], bF[7]));
    uint4 u1 = make_uint4(pk_bf16(bF[8], bF[9]),   pk_bf16(bF[10], bF[11]),
                          pk_bf16(bF[12], bF[13]), pk_bf16(bF[14], bF[15]));
    *(uint4*)&Lb[bn * 40 + (bkh << 4)]     = u0;
    *(uint4*)&Lb[bn * 40 + (bkh << 4) + 8] = u1;
  };

  // prologue: stage tile 0
  load_tiles(0);
  store_tiles(0);

  for (int kt = 0; kt < T; ++kt) {
    __syncthreads();                  // buf (kt&1) now visible to all waves
    const int buf = kt & 1;

    // issue global loads for next tile before consuming current one
    if (kt + 1 < T) load_tiles(kt + 1);

    // L2 prefetch of streaming A operand two tiles ahead
    if (kt + 2 < T) {
      const int k2 = (kt + 2) << 5;
      if constexpr (sizeof(AT) == 4)
        __builtin_prefetch(A + (size_t)(m0 + (tid >> 3)) * K + k2, 0, 2);
      else
        __builtin_prefetch(A + (size_t)(m0 + (tid >> 2)) * K + k2, 0, 2);
    }

    // ---------------- compute: 2x4 WMMA tiles per wave ----------------
    const __bf16* La = la[buf];
    const __bf16* Lb = lb[buf];
    v16bf af[2], bfv[4];
    #pragma unroll
    for (int tm = 0; tm < 2; ++tm) {
      const __bf16* p = &La[(waveM + (tm << 4) + ml) * 40 + khf];
      v8bf lo = *(const v8bf*)p;
      v8bf hi = *(const v8bf*)(p + 16);
      af[tm] = __builtin_shufflevector(lo, hi, 0,1,2,3,4,5,6,7,8,9,10,11,12,13,14,15);
    }
    #pragma unroll
    for (int tn = 0; tn < 4; ++tn) {
      const __bf16* p = &Lb[(waveN + (tn << 4) + ml) * 40 + khf];
      v8bf lo = *(const v8bf*)p;
      v8bf hi = *(const v8bf*)(p + 16);
      bfv[tn] = __builtin_shufflevector(lo, hi, 0,1,2,3,4,5,6,7,8,9,10,11,12,13,14,15);
    }
    #pragma unroll
    for (int tm = 0; tm < 2; ++tm)
      #pragma unroll
      for (int tn = 0; tn < 4; ++tn)
        acc[tm][tn] = __builtin_amdgcn_wmma_f32_16x16x32_bf16(
            false, af[tm], false, bfv[tn], (short)0, acc[tm][tn], false, false);

    // stage next tile into the opposite buffer (safe: readers of this buffer
    // finished before the barrier at the top of this iteration)
    if (kt + 1 < T) store_tiles(buf ^ 1);
  }

  // ---------------- epilogue: bias + activation + store ----------------
  const int mrow = (lane >> 4) << 3;   // 0 or 8
  #pragma unroll
  for (int tm = 0; tm < 2; ++tm) {
    #pragma unroll
    for (int tn = 0; tn < 4; ++tn) {
      const int gn = n0 + waveN + (tn << 4) + ml;
      const float bv = bias[gn];
      #pragma unroll
      for (int r = 0; r < 8; ++r) {
        const int gm = m0 + waveM + (tm << 4) + mrow + r;
        float x = acc[tm][tn][r] + bv;
        if constexpr (ACT == 0) {
          out[(size_t)gm * N + gn] = (OT)gelu_tanh(x);
        } else {
          out[(size_t)gm * N + gn] = (OT)sig2(x);
        }
      }
    }
  }
}

// Per (row, group): top-2 of 64 with lowest-index tie-break (matches lax.top_k).
__global__ __launch_bounds__(256) void topk2_kernel(
    const float* __restrict__ coeff, float* __restrict__ tkc, int* __restrict__ tki)
{
  int t = blockIdx.x * blockDim.x + threadIdx.x;
  if (t >= NROWS * NGRP) return;
  int row = t / NGRP, g = t % NGRP;
  const float* p = coeff + (size_t)row * OUTD + g * POOL;

  float m0 = -3.402823466e38f; int i0 = 0;
  for (int i = 0; i < POOL; ++i) { float v = p[i]; if (v > m0) { m0 = v; i0 = i; } }
  float m1 = -3.402823466e38f; int i1 = 0;
  for (int i = 0; i < POOL; ++i) {
    if (i == i0) continue;
    float v = p[i]; if (v > m1) { m1 = v; i1 = i; }
  }
  size_t o = (size_t)row * (NGRP * TOPK) + g * TOPK;
  tkc[o] = m0; tkc[o + 1] = m1;
  tki[o] = i0; tki[o + 1] = i1;
}

extern "C" void kernel_launch(void* const* d_in, const int* in_sizes, int n_in,
                              void* d_out, int out_size, void* d_ws, size_t ws_size,
                              hipStream_t stream) {
  const float* q  = (const float*)d_in[0];
  const float* W1 = (const float*)d_in[1];
  const float* b1 = (const float*)d_in[2];
  const float* W2 = (const float*)d_in[3];
  const float* b2 = (const float*)d_in[4];
  const float* W3 = (const float*)d_in[5];
  const float* b3 = (const float*)d_in[6];

  __bf16* h1 = (__bf16*)d_ws;                       // 16384*4096 bf16 = 128 MiB
  __bf16* h2 = h1 + (size_t)NROWS * HID;            // next 128 MiB

  float* coeff = (float*)d_out;                                   // [B, 384]
  float* tkc   = coeff + (size_t)NROWS * OUTD;                    // [B, 12]
  int*   tki   = (int*)d_out + (size_t)NROWS * OUTD + (size_t)NROWS * NGRP * TOPK;

  dim3 blk(256);
  // L1: h1 = gelu(q @ W1 + b1)
  gemm_bias_act<float, 0, __bf16>
      <<<dim3(HID / 128, NROWS / 128), blk, 0, stream>>>(q, W1, b1, h1, NROWS, HID, IN_DIM);
  // L2: h2 = gelu(h1 @ W2 + b2)
  gemm_bias_act<__bf16, 0, __bf16>
      <<<dim3(HID / 128, NROWS / 128), blk, 0, stream>>>(h1, W2, b2, h2, NROWS, HID, HID);
  // L3: coeff = 2*sigmoid(h2 @ W3 + b3)
  gemm_bias_act<__bf16, 1, float>
      <<<dim3(OUTD / 128, NROWS / 128), blk, 0, stream>>>(h2, W3, b3, coeff, NROWS, OUTD, HID);
  // top-2 per group
  topk2_kernel<<<(NROWS * NGRP + 255) / 256, blk, 0, stream>>>(coeff, tkc, tki);
}
